// Local_Attention_82291573391387
// MI455X (gfx1250) — matrix operand order repaired and checked by reading, not compiled, not run on hardware
//
#include <hip/hip_runtime.h>
#include <math.h>

typedef __attribute__((ext_vector_type(16))) _Float16 v16h;
typedef __attribute__((ext_vector_type(8)))  float    v8f;

#define LSTRIDE 49          // 48 attn cols padded to odd stride -> conflict-free row reads
#define WAVES_PER_BLOCK 8

// One wave32 handles one 16-query tile for one (b,h).
// Band width w=7 => each 16-query tile's keys live in 3 adjacent 16-key tiles.
__global__ __launch_bounds__(256)
void local_attn_wmma(const float* __restrict__ Q,
                     const float* __restrict__ Kp,
                     const float* __restrict__ Vp,
                     float* __restrict__ Out,
                     int B, int L, int H, int E, int w, float scale)
{
    __shared__ float attnS[WAVES_PER_BLOCK][16 * LSTRIDE];

    const int tid    = threadIdx.x;
    const int wavein = tid >> 5;
    const int lane   = tid & 31;
    const int half   = lane >> 4;   // which 16-lane half of the wave
    const int lr     = lane & 15;

    const int nt   = L >> 4;                       // query/key tiles along L
    const int item = blockIdx.x * WAVES_PER_BLOCK + wavein;
    const int qt   = item % nt;
    const int h    = (item / nt) % H;
    const int b    = item / (nt * H);
    const int q0   = qt << 4;
    const int keybase = (qt - 1) << 4;             // first key of the 3-tile band window

    float* sm = &attnS[wavein][0];

    // ---------------- Q tile -> WMMA A operands (two K=32 chunks of E=64) ------------
    // A operand (16x32): lane holds row M=lr; this code uses element e -> K = e + 16*half (same order as B).
    const float* qrow = Q + ((size_t)(b * L + (q0 + lr)) * H + h) * E;
    v16h aQ0, aQ1;
#pragma unroll
    for (int e = 0; e < 16; ++e) {
        const int k0 = e + 16 * half;   // 0..31
        aQ0[e] = (_Float16)qrow[k0];
        aQ1[e] = (_Float16)qrow[k0 + 32];
    }

    // ---------------- scores = QK^T over 3 key tiles, masked + scaled ----------------
    float sc[3][8];
#pragma unroll
    for (int t = 0; t < 3; ++t) {
        const int kt = qt - 1 + t;
        v8f s = {0.f, 0.f, 0.f, 0.f, 0.f, 0.f, 0.f, 0.f};
        if (kt >= 0 && kt < nt) {                  // wave-uniform branch (EXEC stays full)
            // B operand (32x16): lane holds column N=lr; this code uses element e -> K = e + 16*half (same order as A).
            const float* krow = Kp + ((size_t)(b * L + (kt * 16 + lr)) * H + h) * E;
            v16h bK0, bK1;
#pragma unroll
            for (int e = 0; e < 16; ++e) {
                bK0[e] = (_Float16)krow[e + 16 * half];
                bK1[e] = (_Float16)krow[e + 16 * half + 32];
            }
            s = __builtin_amdgcn_wmma_f32_16x16x32_f16(false, aQ0, false, bK0, (short)0, s, false, false);
            s = __builtin_amdgcn_wmma_f32_16x16x32_f16(false, aQ1, false, bK1, (short)0, s, false, false);
        }
        // C/D layout: element r -> query row M = r + 8*half; lane -> key col N = lr.
#pragma unroll
        for (int r = 0; r < 8; ++r) {
            const int i = q0 + r + 8 * half;
            const int j = kt * 16 + lr;
            int d = i - j; if (d < 0) d = -d;
            const bool valid = (kt >= 0) && (kt < nt) && (d <= w);
            sc[t][r] = valid ? s[r] * scale : -3.0e38f;
        }
    }

    // ---------------- row softmax (reduce across the 16 lanes of this half) ----------
#pragma unroll
    for (int r = 0; r < 8; ++r) {
        float m = fmaxf(fmaxf(sc[0][r], sc[1][r]), sc[2][r]);
#pragma unroll
        for (int off = 1; off < 16; off <<= 1)
            m = fmaxf(m, __shfl_xor(m, off, 32));
        const float p0 = __expf(sc[0][r] - m);
        const float p1 = __expf(sc[1][r] - m);
        const float p2 = __expf(sc[2][r] - m);
        float ssum = p0 + p1 + p2;
#pragma unroll
        for (int off = 1; off < 16; off <<= 1)
            ssum += __shfl_xor(ssum, off, 32);
        const float inv = 1.0f / ssum;
        const int  M   = r + 8 * half;
        sm[M * LSTRIDE + 0 * 16 + lr] = p0 * inv;   // LDS stage: transpose C-layout -> A-layout
        sm[M * LSTRIDE + 1 * 16 + lr] = p1 * inv;
        sm[M * LSTRIDE + 2 * 16 + lr] = p2 * inv;
    }
    __syncthreads();

    // ---------------- attn -> WMMA A operands (48 keys padded to K=64) ---------------
    v16h aA0, aA1;
#pragma unroll
    for (int e = 0; e < 16; ++e) {
        const int k0 = e + 16 * half;   // 0..31
        aA0[e] = (_Float16)sm[lr * LSTRIDE + k0];
        const int k1 = k0 + 32;                                         // 32..63
        aA1[e] = (k1 < 48) ? (_Float16)sm[lr * LSTRIDE + k1] : (_Float16)0.f;
    }

    // ---------------- out = attn @ V, 4 d-tiles of 16 --------------------------------
#pragma unroll
    for (int dt = 0; dt < 4; ++dt) {
        v16h bV0, bV1;
#pragma unroll
        for (int e = 0; e < 16; ++e) {
            const int kd0 = e + 16 * half;              // 0..31
            const int ky0 = keybase + kd0;
            float x0 = 0.f;
            if (ky0 >= 0 && ky0 < L)
                x0 = Vp[((size_t)(b * L + ky0) * H + h) * E + dt * 16 + lr];
            bV0[e] = (_Float16)x0;

            const int kd1 = kd0 + 32;                   // 32..63 (only 32..47 are real)
            float x1 = 0.f;
            if (kd1 < 48) {
                const int ky1 = keybase + kd1;
                if (ky1 >= 0 && ky1 < L)
                    x1 = Vp[((size_t)(b * L + ky1) * H + h) * E + dt * 16 + lr];
            }
            bV1[e] = (_Float16)x1;
        }
        v8f o = {0.f, 0.f, 0.f, 0.f, 0.f, 0.f, 0.f, 0.f};
        o = __builtin_amdgcn_wmma_f32_16x16x32_f16(false, aA0, false, bV0, (short)0, o, false, false);
        o = __builtin_amdgcn_wmma_f32_16x16x32_f16(false, aA1, false, bV1, (short)0, o, false, false);
#pragma unroll
        for (int r = 0; r < 8; ++r) {
            const int row = q0 + r + 8 * half;
            Out[((size_t)(b * L + row) * H + h) * E + dt * 16 + lr] = o[r];
        }
    }
}

extern "C" void kernel_launch(void* const* d_in, const int* in_sizes, int n_in,
                              void* d_out, int out_size, void* d_ws, size_t ws_size,
                              hipStream_t stream) {
    (void)in_sizes; (void)n_in; (void)d_ws; (void)ws_size; (void)out_size;
    // Shapes fixed by setup_inputs(): [B, L, H, E] = [2, 2048, 8, 64], float32.
    const int B = 2, L = 2048, H = 8, E = 64;
    const int w = (int)ceil(1.2 * (log((double)L) / log(2.0)) / 2.0);  // = 7
    const float scale = 1.0f / sqrtf((float)E);

    const float* Q = (const float*)d_in[0];
    const float* K = (const float*)d_in[1];
    const float* V = (const float*)d_in[2];
    float* O = (float*)d_out;

    const int nt = L / 16;                        // 128 query tiles
    const int waves = B * H * nt;                 // 2048 waves
    const int blocks = waves / WAVES_PER_BLOCK;   // 256 blocks x 256 threads (8 wave32s)
    local_attn_wmma<<<blocks, WAVES_PER_BLOCK * 32, 0, stream>>>(Q, K, V, O, B, L, H, E, w, scale);
}
